// LoraInjectedLinearWrapper_69011534512954
// MI455X (gfx1250) — matrix operand order repaired and checked by reading, not compiled, not run on hardware
//
#include <hip/hip_runtime.h>

#define D_IN   4096
#define D_OUT  4096
#define R_RANK 16
#define M_TOTAL (8 * 2048)
#define BM 128
#define BN 64
#define BK 32
#define LSTR 40            // padded LDS row stride (bf16 elements)
#define KSTEPS (D_IN / BK)
#define SCALE 1.0f

typedef __attribute__((ext_vector_type(16))) __bf16         v16bf;
typedef __attribute__((ext_vector_type(8)))  float          v8f;
typedef __attribute__((ext_vector_type(16))) unsigned short v16us;
typedef __attribute__((ext_vector_type(8)))  unsigned short v8us;

static __device__ __forceinline__ unsigned short f2bf(float f) {
  unsigned u = __float_as_uint(f);
  u += 0x7FFFu + ((u >> 16) & 1u);       // round-to-nearest-even
  return (unsigned short)(u >> 16);
}
static __device__ __forceinline__ float bf2f(unsigned short h) {
  return __uint_as_float(((unsigned)h) << 16);
}
static __device__ __forceinline__ unsigned pack2(unsigned short a, unsigned short b) {
  return (unsigned)a | ((unsigned)b << 16);
}

// Build a v16bf WMMA operand from two 8-element LDS chunks (16B each).
static __device__ __forceinline__ v16bf ld_frag(const unsigned short* p, int off) {
  v8us c0 = *(const v8us*)p;
  v8us c1 = *(const v8us*)(p + off);
  v16us f;
#pragma unroll
  for (int i = 0; i < 8; ++i) { f[i] = c0[i]; f[i + 8] = c1[i]; }
  return __builtin_bit_cast(v16bf, f);
}

__global__ __launch_bounds__(256)
void lora_linear_kernel(const float* __restrict__ x, const float* __restrict__ W,
                        const float* __restrict__ bvec, const float* __restrict__ A,
                        const float* __restrict__ Bm, float* __restrict__ out) {
  __shared__ unsigned short XsHi[BM * LSTR], XsLo[BM * LSTR];   // x tile, row m, col k
  __shared__ unsigned short WsHi[BN * LSTR], WsLo[BN * LSTR];   // W tile transposed: [n][k]
  __shared__ unsigned short AsHi[R_RANK * LSTR], AsLo[R_RANK * LSTR]; // A tile transposed: [r][k]
  __shared__ float BmS[BN * R_RANK];                            // Bm tile transposed: [n][r]
  __shared__ float uS[BM * R_RANK];                             // u = x@A for this block

  const int tid  = threadIdx.x;
  const int lane = tid & 31;
  const int ln   = lane & 15;
  const int lh   = lane >> 4;
  const int wid  = tid >> 5;
  const int wm   = wid & 3;    // 4 wave-rows of 32 M each
  const int wn   = wid >> 2;   // 2 wave-cols of 32 N each

  const int n0 = blockIdx.x * BN;   // N fastest -> x slab stays hot in L2, W goes L2-resident
  const int m0 = blockIdx.y * BM;

  // Stage Bm tile transposed (once per block)
  {
    int n = tid >> 2, r0 = (tid & 3) * 4;
    float4 v;
    v.x = Bm[(size_t)(r0 + 0) * D_OUT + n0 + n];
    v.y = Bm[(size_t)(r0 + 1) * D_OUT + n0 + n];
    v.z = Bm[(size_t)(r0 + 2) * D_OUT + n0 + n];
    v.w = Bm[(size_t)(r0 + 3) * D_OUT + n0 + n];
    *(float4*)&BmS[n * R_RANK + r0] = v;
  }

  v8f c[2][2] = {};
  v8f uacc[2] = {};

  const int xm  = tid >> 1;          // x staging: row 0..127
  const int xkb = (tid & 1) * 16;    //            k half
  const int wnn = tid & 63;          // W staging: col
  const int wkb = (tid >> 6) * 8;    //            k base
  const int ak  = tid >> 3;          // A staging: k 0..31
  const int arp = (tid & 7) * 2;     //            r pair

  for (int kk = 0; kk < KSTEPS; ++kk) {
    const int k0 = kk * BK;
    __syncthreads();

    // ---- stage x tile, split into hi/lo bf16 ----
    {
      const float* xp = x + (size_t)(m0 + xm) * D_IN + (k0 + xkb);
#pragma unroll
      for (int i = 0; i < 4; ++i) {
        float4 v = *(const float4*)(xp + i * 4);
        unsigned short h0 = f2bf(v.x), h1 = f2bf(v.y), h2 = f2bf(v.z), h3 = f2bf(v.w);
        unsigned short l0 = f2bf(v.x - bf2f(h0)), l1 = f2bf(v.y - bf2f(h1));
        unsigned short l2 = f2bf(v.z - bf2f(h2)), l3 = f2bf(v.w - bf2f(h3));
        int o = xm * LSTR + xkb + i * 4;
        *(unsigned*)&XsHi[o]     = pack2(h0, h1);
        *(unsigned*)&XsHi[o + 2] = pack2(h2, h3);
        *(unsigned*)&XsLo[o]     = pack2(l0, l1);
        *(unsigned*)&XsLo[o + 2] = pack2(l2, l3);
      }
      if (kk + 1 < KSTEPS) __builtin_prefetch(xp + BK, 0, 0);  // global_prefetch_b8
    }
    // ---- stage W tile transposed [n][k], split hi/lo ----
    {
      const float* wp = W + (size_t)(k0 + wkb) * D_OUT + (n0 + wnn);
#pragma unroll
      for (int j = 0; j < 8; j += 2) {
        float w0 = wp[(size_t)j * D_OUT];
        float w1 = wp[(size_t)(j + 1) * D_OUT];
        unsigned short h0 = f2bf(w0), h1 = f2bf(w1);
        unsigned short l0 = f2bf(w0 - bf2f(h0)), l1 = f2bf(w1 - bf2f(h1));
        int o = wnn * LSTR + wkb + j;
        *(unsigned*)&WsHi[o] = pack2(h0, h1);
        *(unsigned*)&WsLo[o] = pack2(l0, l1);
      }
    }
    // ---- stage A tile transposed [r][k], split hi/lo ----
    {
      const float* ap = A + (size_t)(k0 + ak) * R_RANK + arp;
      float a0 = ap[0], a1 = ap[1];
      unsigned short h0 = f2bf(a0), h1 = f2bf(a1);
      AsHi[arp * LSTR + ak]       = h0;
      AsHi[(arp + 1) * LSTR + ak] = h1;
      AsLo[arp * LSTR + ak]       = f2bf(a0 - bf2f(h0));
      AsLo[(arp + 1) * LSTR + ak] = f2bf(a1 - bf2f(h1));
    }
    __syncthreads();

    // ---- fragment loads (contiguous k = 16*lh + s, same order for A and B) ----
    v16bf ahi[2], alo[2], bhi[2], blo[2];
#pragma unroll
    for (int mt = 0; mt < 2; ++mt) {
      int row = (wm * 32 + mt * 16 + ln) * LSTR + lh * 16; // A: 16 contiguous K per half
      ahi[mt] = ld_frag(&XsHi[row], 8);
      alo[mt] = ld_frag(&XsLo[row], 8);
    }
#pragma unroll
    for (int nt = 0; nt < 2; ++nt) {
      int row = (wn * 32 + nt * 16 + ln) * LSTR + lh * 16; // B: 16 contiguous K per half
      bhi[nt] = ld_frag(&WsHi[row], 8);
      blo[nt] = ld_frag(&WsLo[row], 8);
    }

    // ---- main GEMM: compensated split-bf16 (hi*hi + hi*lo + lo*hi) ----
#pragma unroll
    for (int mt = 0; mt < 2; ++mt)
#pragma unroll
      for (int nt = 0; nt < 2; ++nt) {
        c[mt][nt] = __builtin_amdgcn_wmma_f32_16x16x32_bf16(false, ahi[mt], false, bhi[nt], (short)0, c[mt][nt], false, false);
        c[mt][nt] = __builtin_amdgcn_wmma_f32_16x16x32_bf16(false, ahi[mt], false, blo[nt], (short)0, c[mt][nt], false, false);
        c[mt][nt] = __builtin_amdgcn_wmma_f32_16x16x32_bf16(false, alo[mt], false, bhi[nt], (short)0, c[mt][nt], false, false);
      }

    // ---- LoRA down-proj u += x@A (reuses x fragments; wn==0 waves only) ----
    if (wn == 0) {
      int arow = ln * LSTR + lh * 16;
      v16bf ghi = ld_frag(&AsHi[arow], 8);
      v16bf glo = ld_frag(&AsLo[arow], 8);
#pragma unroll
      for (int mt = 0; mt < 2; ++mt) {
        uacc[mt] = __builtin_amdgcn_wmma_f32_16x16x32_bf16(false, ahi[mt], false, ghi, (short)0, uacc[mt], false, false);
        uacc[mt] = __builtin_amdgcn_wmma_f32_16x16x32_bf16(false, ahi[mt], false, glo, (short)0, uacc[mt], false, false);
        uacc[mt] = __builtin_amdgcn_wmma_f32_16x16x32_bf16(false, alo[mt], false, ghi, (short)0, uacc[mt], false, false);
      }
    }
  }

  // ---- publish u (C layout: m = 8*lh + v, r = ln) ----
  if (wn == 0) {
#pragma unroll
    for (int mt = 0; mt < 2; ++mt)
#pragma unroll
      for (int v = 0; v < 8; ++v) {
        int mloc = wm * 32 + mt * 16 + lh * 8 + v;
        uS[mloc * R_RANK + ln] = uacc[mt][v];
      }
  }
  __syncthreads();

  // ---- tail: out = c + bias + SCALE * u @ Bm ----
  float4 bm[2][4];
  float bias[2];
#pragma unroll
  for (int nt = 0; nt < 2; ++nt) {
    int nidx = wn * 32 + nt * 16 + ln;
    const float4* bp = (const float4*)&BmS[nidx * R_RANK];
    bm[nt][0] = bp[0]; bm[nt][1] = bp[1]; bm[nt][2] = bp[2]; bm[nt][3] = bp[3];
    bias[nt] = bvec[n0 + nidx];
  }
#pragma unroll
  for (int mt = 0; mt < 2; ++mt) {
#pragma unroll
    for (int v = 0; v < 8; ++v) {
      int mloc = wm * 32 + mt * 16 + lh * 8 + v;
      const float4* up = (const float4*)&uS[mloc * R_RANK];
      float4 u0 = up[0], u1 = up[1], u2 = up[2], u3 = up[3];
#pragma unroll
      for (int nt = 0; nt < 2; ++nt) {
        float acc =
          u0.x * bm[nt][0].x + u0.y * bm[nt][0].y + u0.z * bm[nt][0].z + u0.w * bm[nt][0].w +
          u1.x * bm[nt][1].x + u1.y * bm[nt][1].y + u1.z * bm[nt][1].z + u1.w * bm[nt][1].w +
          u2.x * bm[nt][2].x + u2.y * bm[nt][2].y + u2.z * bm[nt][2].z + u2.w * bm[nt][2].w +
          u3.x * bm[nt][3].x + u3.y * bm[nt][3].y + u3.z * bm[nt][3].z + u3.w * bm[nt][3].w;
        int nidx = wn * 32 + nt * 16 + ln;
        out[(size_t)(m0 + mloc) * D_OUT + (n0 + nidx)] = c[mt][nt][v] + bias[nt] + SCALE * acc;
      }
    }
  }
}

extern "C" void kernel_launch(void* const* d_in, const int* in_sizes, int n_in,
                              void* d_out, int out_size, void* d_ws, size_t ws_size,
                              hipStream_t stream) {
  (void)in_sizes; (void)n_in; (void)d_ws; (void)ws_size; (void)out_size;
  const float* x  = (const float*)d_in[0];
  const float* W  = (const float*)d_in[1];
  const float* b  = (const float*)d_in[2];
  const float* A  = (const float*)d_in[3];
  const float* Bm = (const float*)d_in[4];
  float* out = (float*)d_out;
  dim3 grid(D_OUT / BN, M_TOTAL / BM);   // N fastest-varying for L2 reuse of x slab
  dim3 block(256);
  hipLaunchKernelGGL(lora_linear_kernel, grid, block, 0, stream, x, W, b, A, Bm, out);
}
